// Attention_59614146068935
// MI455X (gfx1250) — matrix operand order repaired and checked by reading, not compiled, not run on hardware
//
#include <hip/hip_runtime.h>

// MI455X / gfx1250, wave32. WMMA bf16 16x16x32, f32 accumulate; TDM for the
// bulk score-tile copy into LDS in kernel 2.
typedef __attribute__((ext_vector_type(16))) __bf16 v16bf;
typedef __attribute__((ext_vector_type(8)))  float  v8f;
typedef __attribute__((ext_vector_type(4)))  float  float4v;

#define B_  2
#define H_  16
#define NQ_ 1024
#define NK_ 1024
#define DH_ 128
#define DD_ 8

#if defined(__has_builtin)
# if __has_builtin(__builtin_amdgcn_tensor_load_to_lds) && \
     __has_builtin(__builtin_amdgcn_s_wait_tensorcnt)
#  define USE_TDM 1
# endif
#endif
#ifndef USE_TDM
# define USE_TDM 0
#endif

// ---------------------------------------------------------------------------
// Kernel 1: modulated scores.
// Block = 256 threads (8 waves). Each block: one (b, h, 16-row q tile).
// Each wave covers 128 keys (8 key tiles of 16). Writes RAW modulated scores
// into the p-region of d_out (staging; kernel 2 normalizes in place).
// ---------------------------------------------------------------------------
__global__ void __launch_bounds__(256) score_kernel(
    const float* __restrict__ q,  const float* __restrict__ kmat,
    const float* __restrict__ dq, const float* __restrict__ dtop,
    const float* __restrict__ dbot, const float* __restrict__ dscore,
    const float* __restrict__ ww, const float* __restrict__ bw,
    const float* __restrict__ wb, const float* __restrict__ bb,
    float* __restrict__ sc)
{
  const int tid  = threadIdx.x;
  const int wave = tid >> 5, lane = tid & 31;
  const int bid  = blockIdx.x;
  const int qt   = bid & 63;          // NQ/16
  const int h    = (bid >> 6) & 15;
  const int b    = bid >> 10;
  const int qbase = qt << 4;
  const int row16 = lane & 15;
  const int hi    = lane >> 4;
  const int hi8   = hi << 3;
  const int hi16  = hi << 4;

  // per-head projection columns (tiny, uniform across lanes)
  float wreg[8], breg[8];
#pragma unroll
  for (int d = 0; d < 8; ++d) { wreg[d] = ww[d*H_ + h]; breg[d] = wb[d*H_ + h]; }
  const float bwh = bw[h], bbh = bb[h];

  // per-row gate terms A_w[q], A_b[q] for the 8 rows this lane's C/D covers
  float Aw[8], Ab[8];
#pragma unroll
  for (int r = 0; r < 8; ++r) {
    const float* dqp = dq + ((size_t)b*NQ_ + qbase + r + hi8)*DD_;
    float aw = 0.f, ab = 0.f;
#pragma unroll
    for (int d = 0; d < 8; ++d) { aw += dqp[d]*wreg[d]; ab += dqp[d]*breg[d]; }
    Aw[r] = aw; Ab[r] = ab;
  }

  // Q fragments (A operand), 4 chunks of K=32 over DH=128; two 8-float runs
  const float* qrow = q + (((size_t)b*H_ + h)*NQ_ + qbase + row16)*DH_;
  v16bf qa[4];
#pragma unroll
  for (int c = 0; c < 4; ++c) {
#pragma unroll
    for (int run = 0; run < 2; ++run) {
      const float* p0 = qrow + c*32 + run*16 + hi8;
#pragma unroll
      for (int i = 0; i < 8; ++i) qa[c][run*8 + i] = (__bf16)p0[i];
    }
  }

  const float scale = 0.08838834764831845f; // 1/sqrt(128)

  for (int kt8 = 0; kt8 < 8; ++kt8) {
    const int kcol = (wave*8 + kt8)*16 + row16;   // this lane's key column
    // per-key gate scalars: sSum, C_w, C_b
    const float s0 = dscore[((size_t)b*NK_ + kcol)*2 + 0];
    const float s1 = dscore[((size_t)b*NK_ + kcol)*2 + 1];
    const float ssum = s0 + s1;
    const float* tp = dtop + ((size_t)b*NK_ + kcol)*DD_;
    const float* bp = dbot + ((size_t)b*NK_ + kcol)*DD_;
    float Cw = 0.f, Cb = 0.f;
#pragma unroll
    for (int d = 0; d < 8; ++d) {
      const float e = tp[d]*s0 + bp[d]*s1;
      Cw += e*wreg[d]; Cb += e*breg[d];
    }

    // scores tile = Q * K^T over DH in 4 chunks of 32 (bf16 WMMA)
    v8f acc = {0.f,0.f,0.f,0.f,0.f,0.f,0.f,0.f};
    const float* krow = kmat + (((size_t)b*H_ + h)*NK_ + kcol)*DH_;
    if (kt8 < 7) __builtin_prefetch(krow + 16*DH_, 0, 3);  // next key tile
#pragma unroll
    for (int c = 0; c < 4; ++c) {
      v16bf kb;
#pragma unroll
      for (int run = 0; run < 2; ++run) {
        const float* p0 = krow + c*32 + run*16 + hi8;   // B: two 8-float runs (table order)
#pragma unroll
        for (int i = 0; i < 8; ++i) kb[run*8 + i] = (__bf16)p0[i];
      }
      acc = __builtin_amdgcn_wmma_f32_16x16x32_bf16(false, qa[c], false, kb,
                                                    (short)0, acc, false, false);
    }

    // modulate: s*softplus(ssum*Aw - Cw + bwh) + (ssum*Ab - Cb + bbh); store raw
    float* srow = sc + ((((size_t)b*H_ + h)*NQ_ + qbase + hi8)*NK_) + kcol;
#pragma unroll
    for (int r = 0; r < 8; ++r) {
      const float t  = ssum*Aw[r] - Cw + bwh;
      const float sp = fmaxf(t, 0.f) + __logf(1.f + __expf(-fabsf(t)));
      const float db = ssum*Ab[r] - Cb + bbh;
      srow[(size_t)r*NK_] = acc[r]*scale*sp + db;
    }
  }
}

// ---------------------------------------------------------------------------
// Kernel 2: TDM tile fetch + softmax (64KB LDS) + p write + P@V with WMMA.
// Block = 128 threads (4 waves). One (b, h, 16-row q tile) per block.
// 1/l is folded into the final accumulator scale (out) and the p store.
// ---------------------------------------------------------------------------
__global__ void __launch_bounds__(128) softmax_pv_kernel(
    const float* __restrict__ v, float* __restrict__ p, float* __restrict__ out)
{
  extern __shared__ char smem[];
  float*  sS    = (float*)smem;                               // 16 x 1024 f32 (64KB)
  __bf16* sV    = (__bf16*)(smem + 16*NK_*4);                 // 128 x 32 bf16 (8KB)
  float*  sLinv = (float*)(smem + 16*NK_*4 + DH_*32*2);       // 16 f32

  const int tid  = threadIdx.x;
  const int wave = tid >> 5, lane = tid & 31;
  const int bid  = blockIdx.x;
  const int qt   = bid & 63;
  const int h    = (bid >> 6) & 15;
  const int b    = bid >> 10;
  const int qbase = qt << 4;
  const int row16 = lane & 15;
  const int hi = lane >> 4, hi8 = hi << 3, hi16 = hi << 4;

  float* pBase = p + (((size_t)b*H_ + h)*NQ_ + qbase)*NK_;    // 16x1024 tile

  // 1) pull the contiguous 64KB raw-score tile into LDS.
#if USE_TDM
  // Tensor Data Mover: one 1-D descriptor (16384 f32), issued by wave 0 only
  // (TDM ignores EXEC; one instruction moves the whole tile, SIMDs stay free).
  if (wave == 0) {
    typedef unsigned int v4u __attribute__((ext_vector_type(4)));
    typedef int          v8i __attribute__((ext_vector_type(8)));
    typedef int          v4i __attribute__((ext_vector_type(4)));
    const unsigned long long ga = (unsigned long long)(uintptr_t)pBase;
    const unsigned int ldsoff   = (unsigned int)(uintptr_t)sS; // LDS byte offset
    v4u g0; v8i g1;
    v4i gz4 = {0, 0, 0, 0};
    v8i gz8 = {0, 0, 0, 0, 0, 0, 0, 0};
    // D# group 0: count=1 | lds_addr | global_addr[56:0] | type=2
    g0[0] = 1u;
    g0[1] = ldsoff;
    g0[2] = (unsigned int)ga;
    g0[3] = (unsigned int)((ga >> 32) & 0x01FFFFFFull) | (2u << 30);
    // D# group 1: data_size=4B; tensor_dim0=tile_dim0=stride0=16384; 1-D tile
    g1[0] = (int)(2u << 16);           // data_size code 2 (4 bytes)
    g1[1] = (int)(16384u << 16);       // tensor_dim0[15:0]
    g1[2] = (int)(1u << 16);           // tensor_dim0[31:16]=0 | tensor_dim1=1
    g1[3] = (int)(16384u << 16);       // tile_dim0 = 16384
    g1[4] = 0;                         // tile_dim1/tile_dim2 unused
    g1[5] = 16384;                     // tensor_dim0_stride[31:0]
    g1[6] = 0;
    g1[7] = 0;
    __builtin_amdgcn_tensor_load_to_lds(g0, g1, gz4, gz4, gz8, 0);
    __builtin_amdgcn_s_wait_tensorcnt(0);
  }
  __syncthreads();
#else
#pragma unroll
  for (int i = 0; i < 32; ++i) {
    const int f4i = tid + i*128;            // 4096 float4 units
    ((float4v*)sS)[f4i] = ((const float4v*)pBase)[f4i];
  }
  __syncthreads();
#endif

  // 2) row softmax stats (8 lanes per row); exp(s-m) kept in LDS; 1/l saved
  {
    const int row = tid >> 3, sub = tid & 7;
    float m = -3.0e38f;
    for (int j = 0; j < 128; ++j) m = fmaxf(m, sS[row*NK_ + sub + j*8]);
#pragma unroll
    for (int x = 1; x < 8; x <<= 1) m = fmaxf(m, __shfl_xor(m, x, 32));
    float l = 0.f;
    for (int j = 0; j < 128; ++j) {
      const int idx = row*NK_ + sub + j*8;
      const float e = __expf(sS[idx] - m);
      sS[idx] = e;
      l += e;
    }
#pragma unroll
    for (int x = 1; x < 8; x <<= 1) l += __shfl_xor(l, x, 32);
    if (sub == 0) sLinv[row] = 1.f / l;
  }
  __syncthreads();

  float rl[8];
#pragma unroll
  for (int r = 0; r < 8; ++r) rl[r] = sLinv[r + hi8];

  // 3) write normalized p back over the staging region (coalesced float4)
#pragma unroll
  for (int i = 0; i < 32; ++i) {
    const int f4i = tid + i*128;
    const int row = f4i >> 8;               // 256 float4 per row
    float4v pv = ((float4v*)sS)[f4i];
    pv *= sLinv[row];
    ((float4v*)pBase)[f4i] = pv;
  }

  // 4) out = (p~ @ V) * (1/l): 32 K-chunks of 32; V staged transposed in LDS
  v8f acc0 = {0.f,0.f,0.f,0.f,0.f,0.f,0.f,0.f};
  v8f acc1 = acc0;
  const int n0 = wave << 5;                 // this wave's 32 output columns
  const float* vbase = v + (((size_t)b*H_ + h)*NK_)*DH_;

  for (int kc = 0; kc < 32; ++kc) {
    const int k0 = kc << 5;
    __syncthreads();                        // sV reuse across iterations
    {
      const int kk = tid >> 2;
      const int dbase = (tid & 3) << 5;
      const float* vg = vbase + (size_t)(k0 + kk)*DH_ + dbase;
      if (kc < 31) __builtin_prefetch(vg + 32*DH_, 0, 3);   // next V chunk
#pragma unroll
      for (int i = 0; i < 32; ++i) sV[(dbase + i)*32 + kk] = (__bf16)vg[i];
    }
    __syncthreads();

    // A fragment: p~ rows from LDS (two 8-float runs), f32 -> bf16
    v16bf af;
#pragma unroll
    for (int run = 0; run < 2; ++run) {
      const float* p0 = sS + row16*NK_ + k0 + hi16 + run*8;
#pragma unroll
      for (int i = 0; i < 8; ++i) af[run*8 + i] = (__bf16)p0[i];
    }
    // B fragments: 16 contiguous bf16 per lane-half from transposed V
    v16bf bf0, bf1;
    {
      const __bf16* q0 = sV + (n0 + row16)*32 + hi16;
      const __bf16* q1 = sV + (n0 + 16 + row16)*32 + hi16;
#pragma unroll
      for (int i = 0; i < 16; ++i) { bf0[i] = q0[i]; bf1[i] = q1[i]; }
    }
    acc0 = __builtin_amdgcn_wmma_f32_16x16x32_bf16(false, af, false, bf0,
                                                   (short)0, acc0, false, false);
    acc1 = __builtin_amdgcn_wmma_f32_16x16x32_bf16(false, af, false, bf1,
                                                   (short)0, acc1, false, false);
  }

  // 5) store out, scaled per-row by 1/l
  float* obase = out + (((size_t)b*H_ + h)*NQ_ + qbase + hi8)*DH_;
#pragma unroll
  for (int r = 0; r < 8; ++r) {
    obase[(size_t)r*DH_ + n0 + row16]      = acc0[r]*rl[r];
    obase[(size_t)r*DH_ + n0 + 16 + row16] = acc1[r]*rl[r];
  }
}

// ---------------------------------------------------------------------------
extern "C" void kernel_launch(void* const* d_in, const int* in_sizes, int n_in,
                              void* d_out, int out_size, void* d_ws, size_t ws_size,
                              hipStream_t stream) {
  (void)in_sizes; (void)n_in; (void)d_ws; (void)ws_size; (void)out_size;
  const float* q      = (const float*)d_in[0];
  const float* k      = (const float*)d_in[1];
  const float* v      = (const float*)d_in[2];
  // d_in[3] = c (unused, non-leaky path)
  const float* dq     = (const float*)d_in[4];
  const float* dtop   = (const float*)d_in[5];
  const float* dbot   = (const float*)d_in[6];
  const float* dscore = (const float*)d_in[7];
  const float* ww     = (const float*)d_in[8];
  const float* bw     = (const float*)d_in[9];
  const float* wb     = (const float*)d_in[10];
  const float* bb     = (const float*)d_in[11];

  float* out = (float*)d_out;                       // [B,H,NQ,DH]
  float* p   = out + (size_t)B_*H_*NQ_*DH_;         // [B,H,NQ,NK] (also staging)

  const int nblk = B_*H_*(NQ_/16);                  // 2048
  score_kernel<<<nblk, 256, 0, stream>>>(q, k, dq, dtop, dbot, dscore,
                                         ww, bw, wb, bb, p);
  const size_t lds = (size_t)16*NK_*4 + (size_t)DH_*32*2 + 16*4; // 73792 B
  softmax_pv_kernel<<<nblk, 128, lds, stream>>>(v, p, out);
}
